// PredictiveViTAttention_44968307589375
// MI455X (gfx1250) — compile-verified, matrix operand order checked by reading, not run on hardware
//
#include <hip/hip_runtime.h>
#include <hip/hip_bf16.h>

typedef __attribute__((ext_vector_type(16))) _Float16 v16h;
typedef __attribute__((ext_vector_type(8)))  _Float16 v8h;
typedef __attribute__((ext_vector_type(4)))  _Float16 v4h;
typedef __attribute__((ext_vector_type(8)))  float    v8f;
typedef __attribute__((ext_vector_type(4)))  float    v4f;
typedef __attribute__((ext_vector_type(4)))  unsigned u32x4;
typedef __attribute__((ext_vector_type(8)))  unsigned u32x8;

#define BB 2
#define NN 4096
#define HH 12
#define DD 768
#define HD 64
#define BH (BB*HH)   // 24
#define BN (BB*NN)   // 8192

union V16U { v16h v; v8h h[2]; };

__device__ __forceinline__ v8f wmma_f16(v16h a, v16h b, v8f c) {
  return __builtin_amdgcn_wmma_f32_16x16x32_f16(
      false, a, false, b, (short)0, c, false, false);
}

__device__ __forceinline__ v8f vzero8() {
  v8f z;
#pragma unroll
  for (int i = 0; i < 8; ++i) z[i] = 0.0f;
  return z;
}

// ---------------------------------------------------------------------------
// TDM: 2D tile (64 x 64 f16) global -> LDS with hardware row padding:
// pad_interval = 32 DWORDs (one 128B row), pad_amount = 8 DWORDs (32B)
// => LDS row stride 160B, matching the [..][80] _Float16 arrays below.
// Issued by one wave; EXEC ignored by TDM; tracked with TENSORcnt.
// ---------------------------------------------------------------------------
__device__ __forceinline__ void tdm_load_tile64x64(unsigned ldsOff, const void* gptr,
                                                   unsigned rowStrideElems) {
  unsigned long long ga = (unsigned long long)(size_t)gptr;
  u32x4 g0;
  g0[0] = 1u;                                              // count=1 (valid), user mode
  g0[1] = ldsOff;                                          // lds_addr
  g0[2] = (unsigned)ga;                                    // global_addr[31:0]
  g0[3] = (unsigned)((ga >> 32) & 0x1FFFFFFu) | 0x80000000u; // addr[56:32] | type=2
  u32x8 g1;
  g1[0] = (1u << 16)        // data_size = 2 bytes
        | (1u << 20)        // pad_enable
        | (4u << 22)        // pad_interval = 32 DWORDs
        | (7u << 25);       // pad_amount   = 8 DWORDs
  g1[1] = 64u << 16;        // tensor_dim0 = 64 (low 16 bits at [63:48])
  g1[2] = 64u << 16;        // tensor_dim1 = 64 (low 16 bits at [95:80])
  g1[3] = 64u << 16;        // tile_dim0 = 64 at [127:112]
  g1[4] = 64u;              // tile_dim1 = 64, tile_dim2 = 0
  g1[5] = rowStrideElems;   // tensor_dim0_stride low 32 (data_size units)
  g1[6] = 0u;
  g1[7] = 0u;
#pragma unroll
  for (int i = 0; i < 4; ++i) g0[i] = (unsigned)__builtin_amdgcn_readfirstlane((int)g0[i]);
#pragma unroll
  for (int i = 0; i < 8; ++i) g1[i] = (unsigned)__builtin_amdgcn_readfirstlane((int)g1[i]);
  asm volatile("tensor_load_to_lds %0, %1" :: "s"(g0), "s"(g1) : "memory");
}

// ---------------------------------------------------------------------------
// Kernel 1: fused QKV projection.  C[8192,768] = X * W^T + bias, per matrix.
// Q,K stored f16 as [b][h][n][d]; V stored f16 transposed as [b][h][d][n].
// ---------------------------------------------------------------------------
__global__ __launch_bounds__(256)
void qkv_gemm_kernel(const float* __restrict__ X,
                     const float* __restrict__ Wq, const float* __restrict__ Wk,
                     const float* __restrict__ Wv,
                     const float* __restrict__ bq, const float* __restrict__ bk,
                     const float* __restrict__ bv,
                     _Float16* __restrict__ q_ws, _Float16* __restrict__ k_ws,
                     _Float16* __restrict__ v_ws)
{
  __shared__ _Float16 sA[128][48];  // 96B row stride (16B aligned)
  __shared__ _Float16 sB[64][48];

  const int tid     = threadIdx.x;
  const int rowBase = blockIdx.x * 128;
  const int colBase = blockIdx.y * 64;
  const int mat     = blockIdx.z;
  const float* W    = (mat == 0) ? Wq : ((mat == 1) ? Wk : Wv);
  const float* bias = (mat == 0) ? bq : ((mat == 1) ? bk : bv);

  const int wave = tid >> 5, lane = tid & 31;
  const int nl = lane & 15, hi = lane >> 4, kb = hi * 8;
  const int waveRow = wave * 16;

  v8f cacc[4];
#pragma unroll
  for (int t = 0; t < 4; ++t) cacc[t] = vzero8();

  for (int kk = 0; kk < DD; kk += 32) {
    __syncthreads();
#pragma unroll
    for (int i = 0; i < 4; ++i) {   // A tile 128x32 f32->f16
      int c = tid + i * 256;
      int row = c >> 3, col4 = (c & 7) * 4;
      v4f f = *(const v4f*)(X + (size_t)(rowBase + row) * DD + kk + col4);
      v4h hv; hv[0] = (_Float16)f.x; hv[1] = (_Float16)f.y;
      hv[2] = (_Float16)f.z; hv[3] = (_Float16)f.w;
      *(v4h*)&sA[row][col4] = hv;
    }
#pragma unroll
    for (int i = 0; i < 2; ++i) {   // B tile 64x32 f32->f16
      int c = tid + i * 256;
      int row = c >> 3, col4 = (c & 7) * 4;
      v4f f = *(const v4f*)(W + (size_t)(colBase + row) * DD + kk + col4);
      v4h hv; hv[0] = (_Float16)f.x; hv[1] = (_Float16)f.y;
      hv[2] = (_Float16)f.z; hv[3] = (_Float16)f.w;
      *(v4h*)&sB[row][col4] = hv;
    }
    __syncthreads();

    V16U a, bf[4];
    a.h[0] = *(const v8h*)&sA[waveRow + nl][kb];
    a.h[1] = *(const v8h*)&sA[waveRow + nl][kb + 16];
#pragma unroll
    for (int t = 0; t < 4; ++t) {
      bf[t].h[0] = *(const v8h*)&sB[t * 16 + nl][kb];
      bf[t].h[1] = *(const v8h*)&sB[t * 16 + nl][kb + 16];
    }
#pragma unroll
    for (int t = 0; t < 4; ++t) cacc[t] = wmma_f16(a.v, bf[t].v, cacc[t]);
  }

  const int mrowBase = rowBase + waveRow + hi * 8;   // 8 consecutive tokens
  const int b0  = mrowBase >> 12;
  const int tok = mrowBase & (NN - 1);
#pragma unroll
  for (int t = 0; t < 4; ++t) {
    int col = colBase + t * 16 + nl;
    float bsv = bias[col];
    int hh = col >> 6, dd = col & 63;
    if (mat == 2) {
      v8h pk;
#pragma unroll
      for (int r = 0; r < 8; ++r) pk[r] = (_Float16)(cacc[t][r] + bsv);
      *(v8h*)(v_ws + ((size_t)(b0 * HH + hh) * HD + dd) * NN + tok) = pk;
    } else {
      _Float16* outp = (mat == 0) ? q_ws : k_ws;
#pragma unroll
      for (int r = 0; r < 8; ++r)
        outp[((size_t)(b0 * HH + hh) * NN + (tok + r)) * HD + dd] =
            (_Float16)(cacc[t][r] + bsv);
    }
  }
}

// ---------------------------------------------------------------------------
// Kernel 2: flash attention.  4 waves/block, 16 query rows/wave.
// K/V tiles moved global->LDS by the Tensor Data Mover (one wave posts two
// 2-D descriptors per tile step), with TDM hardware padding giving the
// bank-friendly 160B LDS row stride.
// ---------------------------------------------------------------------------
__global__ __launch_bounds__(128)
void attn_kernel(const _Float16* __restrict__ q_ws, const _Float16* __restrict__ k_ws,
                 const _Float16* __restrict__ v_ws, const unsigned char* __restrict__ mask,
                 _Float16* __restrict__ ctx_ws)
{
  __shared__ _Float16 sK[64][80];       // keys  [m][d]
  __shared__ _Float16 sV[64][80];       // V^T   [d][m]
  __shared__ _Float16 sP[4][16][80];    // per-wave probs [qrow][key]

  const int tid  = threadIdx.x;
  const int bh   = blockIdx.y;
  const int b    = bh / HH, h = bh % HH;
  const int wave = tid >> 5, lane = tid & 31;
  const int nl = lane & 15, hi = lane >> 4, kb = hi * 8;
  const int qbase = blockIdx.x * 64 + wave * 16;

  const unsigned ldsK = (unsigned)(size_t)&sK[0][0];
  const unsigned ldsV = (unsigned)(size_t)&sV[0][0];
  const _Float16* kBase = k_ws + (size_t)bh * NN * HD;
  const _Float16* vBase = v_ws + (size_t)bh * HD * NN;

  // Q A-fragments (two 16x32 frags covering d=0..63)
  const _Float16* qrow = q_ws + ((size_t)bh * NN + qbase + nl) * HD;
  V16U aq0, aq1;
  aq0.h[0] = *(const v8h*)(qrow + kb);       aq0.h[1] = *(const v8h*)(qrow + kb + 16);
  aq1.h[0] = *(const v8h*)(qrow + 32 + kb);  aq1.h[1] = *(const v8h*)(qrow + 48 + kb);

  float Mr[8], Lr[8];
  v8f acc[4];
#pragma unroll
  for (int r = 0; r < 8; ++r) { Mr[r] = -__builtin_inff(); Lr[r] = 0.0f; }
#pragma unroll
  for (int t = 0; t < 4; ++t) acc[t] = vzero8();

  const size_t maskBase = (size_t)b * NN * NN;
  const int qrow0 = qbase + hi * 8;

  for (int kt = 0; kt < NN; kt += 64) {
    __syncthreads();                       // everyone done with previous tiles
    if (wave == 0) {
      tdm_load_tile64x64(ldsK, kBase + (size_t)kt * HD, HD);   // K rows, stride 64
      tdm_load_tile64x64(ldsV, vBase + kt, NN);                // V^T rows, stride 4096
      __builtin_amdgcn_s_wait_tensorcnt(0);
    }
    __syncthreads();                       // tiles visible to all waves

    // prefetch next tiles into L2 while we compute on this one
    if (kt + 64 < NN) {
      __builtin_prefetch(kBase + (size_t)(kt + 64) * HD, 0, 0);
      __builtin_prefetch(vBase + (kt + 64), 0, 0);
    }

    // scores S = Q K^T
    v8f s[4];
#pragma unroll
    for (int t = 0; t < 4; ++t) {
      s[t] = vzero8();
      V16U bk0, bk1;
      bk0.h[0] = *(const v8h*)&sK[t * 16 + nl][kb];
      bk0.h[1] = *(const v8h*)&sK[t * 16 + nl][kb + 16];
      bk1.h[0] = *(const v8h*)&sK[t * 16 + nl][32 + kb];
      bk1.h[1] = *(const v8h*)&sK[t * 16 + nl][48 + kb];
      s[t] = wmma_f16(aq0.v, bk0.v, s[t]);
      s[t] = wmma_f16(aq1.v, bk1.v, s[t]);
    }

    // scale (1/sqrt(64)) + mask
#pragma unroll
    for (int t = 0; t < 4; ++t) {
#pragma unroll
      for (int r = 0; r < 8; ++r) {
        unsigned char mv =
            mask[maskBase + (size_t)(qrow0 + r) * NN + (kt + t * 16 + nl)];
        s[t][r] = mv ? s[t][r] * 0.125f : -__builtin_inff();
      }
    }

    // row max across the 16 lanes holding each row
    float mx[8];
#pragma unroll
    for (int r = 0; r < 8; ++r) {
      float m = fmaxf(fmaxf(s[0][r], s[1][r]), fmaxf(s[2][r], s[3][r]));
      m = fmaxf(m, __shfl_xor(m, 1, 32));
      m = fmaxf(m, __shfl_xor(m, 2, 32));
      m = fmaxf(m, __shfl_xor(m, 4, 32));
      m = fmaxf(m, __shfl_xor(m, 8, 32));
      mx[r] = m;
    }

    float corr[8], rs[8];
#pragma unroll
    for (int r = 0; r < 8; ++r) {
      float Mn = fmaxf(Mr[r], mx[r]);
      corr[r] = __expf(Mr[r] - Mn);
      Mr[r] = Mn;
      rs[r] = 0.0f;
    }

    // p = exp(s - M); row-sum; stash P in per-wave LDS (C-layout -> A-layout)
#pragma unroll
    for (int t = 0; t < 4; ++t) {
#pragma unroll
      for (int r = 0; r < 8; ++r) {
        float p = __expf(s[t][r] - Mr[r]);
        rs[r] += p;
        sP[wave][hi * 8 + r][t * 16 + nl] = (_Float16)p;
      }
    }
#pragma unroll
    for (int r = 0; r < 8; ++r) {
      float t0 = rs[r];
      t0 += __shfl_xor(t0, 1, 32);
      t0 += __shfl_xor(t0, 2, 32);
      t0 += __shfl_xor(t0, 4, 32);
      t0 += __shfl_xor(t0, 8, 32);
      Lr[r] = Lr[r] * corr[r] + t0;
    }
#pragma unroll
    for (int t = 0; t < 4; ++t)
#pragma unroll
      for (int r = 0; r < 8; ++r) acc[t][r] *= corr[r];

    // same-wave LDS RAW on sP
    asm volatile("s_wait_dscnt 0" ::: "memory");

    V16U ap0, ap1;
    ap0.h[0] = *(const v8h*)&sP[wave][nl][kb];
    ap0.h[1] = *(const v8h*)&sP[wave][nl][kb + 16];
    ap1.h[0] = *(const v8h*)&sP[wave][nl][32 + kb];
    ap1.h[1] = *(const v8h*)&sP[wave][nl][48 + kb];

#pragma unroll
    for (int t = 0; t < 4; ++t) {
      V16U bv0, bv1;
      bv0.h[0] = *(const v8h*)&sV[t * 16 + nl][kb];
      bv0.h[1] = *(const v8h*)&sV[t * 16 + nl][kb + 16];
      bv1.h[0] = *(const v8h*)&sV[t * 16 + nl][32 + kb];
      bv1.h[1] = *(const v8h*)&sV[t * 16 + nl][48 + kb];
      acc[t] = wmma_f16(ap0.v, bv0.v, acc[t]);
      acc[t] = wmma_f16(ap1.v, bv1.v, acc[t]);
    }
  }

  // normalize and store ctx as [b][n][h][d] f16
#pragma unroll
  for (int t = 0; t < 4; ++t) {
#pragma unroll
    for (int r = 0; r < 8; ++r) {
      float v = acc[t][r] / Lr[r];
      int q = qrow0 + r;
      ctx_ws[(((size_t)b * NN + q) * HH + h) * HD + (t * 16 + nl)] = (_Float16)v;
    }
  }
}

// ---------------------------------------------------------------------------
// Kernel 3: output projection.  Out[8192,768] = Ctx(f16) * Wo^T + bo (f32).
// A tile is a raw f16 copy -> use async global->LDS loads (ASYNCcnt).
// ---------------------------------------------------------------------------
__global__ __launch_bounds__(256)
void out_gemm_kernel(const _Float16* __restrict__ Ctx, const float* __restrict__ Wo,
                     const float* __restrict__ bo, float* __restrict__ Out)
{
  __shared__ _Float16 sA[128][48];
  __shared__ _Float16 sB[64][48];

  const int tid = threadIdx.x;
  const int rowBase = blockIdx.x * 128;
  const int colBase = blockIdx.y * 64;
  const int wave = tid >> 5, lane = tid & 31;
  const int nl = lane & 15, hi = lane >> 4, kb = hi * 8;
  const int waveRow = wave * 16;

  v8f cacc[4];
#pragma unroll
  for (int t = 0; t < 4; ++t) cacc[t] = vzero8();

  for (int kk = 0; kk < DD; kk += 32) {
    __syncthreads();
    // A tile 128x32 f16: async DMA straight into LDS
#pragma unroll
    for (int i = 0; i < 2; ++i) {
      int c = tid + i * 256;
      int row = c >> 2, col8 = (c & 3) * 8;
      unsigned dst = (unsigned)(size_t)&sA[row][col8];
      unsigned long long src =
          (unsigned long long)(size_t)(Ctx + (size_t)(rowBase + row) * DD + kk + col8);
      asm volatile("global_load_async_to_lds_b128 %0, %1, off"
                   :: "v"(dst), "v"(src) : "memory");
    }
    // B tile 64x32 f32->f16 (conversion must go through VGPRs)
#pragma unroll
    for (int i = 0; i < 2; ++i) {
      int c = tid + i * 256;
      int row = c >> 3, col4 = (c & 7) * 4;
      v4f f = *(const v4f*)(Wo + (size_t)(colBase + row) * DD + kk + col4);
      v4h hv; hv[0] = (_Float16)f.x; hv[1] = (_Float16)f.y;
      hv[2] = (_Float16)f.z; hv[3] = (_Float16)f.w;
      *(v4h*)&sB[row][col4] = hv;
    }
    asm volatile("s_wait_asynccnt 0" ::: "memory");
    __syncthreads();

    V16U a, bf[4];
    a.h[0] = *(const v8h*)&sA[waveRow + nl][kb];
    a.h[1] = *(const v8h*)&sA[waveRow + nl][kb + 16];
#pragma unroll
    for (int t = 0; t < 4; ++t) {
      bf[t].h[0] = *(const v8h*)&sB[t * 16 + nl][kb];
      bf[t].h[1] = *(const v8h*)&sB[t * 16 + nl][kb + 16];
    }
#pragma unroll
    for (int t = 0; t < 4; ++t) cacc[t] = wmma_f16(a.v, bf[t].v, cacc[t]);
  }

  const int mrowBase = rowBase + waveRow + hi * 8;
#pragma unroll
  for (int t = 0; t < 4; ++t) {
    int col = colBase + t * 16 + nl;
    float bsv = bo[col];
#pragma unroll
    for (int r = 0; r < 8; ++r)
      Out[(size_t)(mrowBase + r) * DD + col] = cacc[t][r] + bsv;
  }
}

// ---------------------------------------------------------------------------
extern "C" void kernel_launch(void* const* d_in, const int* in_sizes, int n_in,
                              void* d_out, int out_size, void* d_ws, size_t ws_size,
                              hipStream_t stream) {
  const float*         hs   = (const float*)d_in[0];
  const unsigned char* mask = (const unsigned char*)d_in[1];
  const float* Wq = (const float*)d_in[2];
  const float* bq = (const float*)d_in[3];
  const float* Wk = (const float*)d_in[4];
  const float* bk = (const float*)d_in[5];
  const float* Wv = (const float*)d_in[6];
  const float* bv = (const float*)d_in[7];
  const float* Wo = (const float*)d_in[8];
  const float* bo = (const float*)d_in[9];
  float* out = (float*)d_out;

  const size_t perMat = (size_t)BH * NN * HD;
  _Float16* q_ws   = (_Float16*)d_ws;
  _Float16* k_ws   = q_ws + perMat;
  _Float16* v_ws   = k_ws + perMat;             // transposed [bh][d][n]
  _Float16* ctx_ws = v_ws + perMat;             // [b][n][h*64+d]

  qkv_gemm_kernel<<<dim3(BN / 128, DD / 64, 3), 256, 0, stream>>>(
      hs, Wq, Wk, Wv, bq, bk, bv, q_ws, k_ws, v_ws);
  attn_kernel<<<dim3(NN / 64, BH), 128, 0, stream>>>(
      q_ws, k_ws, v_ws, mask, ctx_ws);
  out_gemm_kernel<<<dim3(BN / 128, DD / 64), 256, 0, stream>>>(
      ctx_ws, Wo, bo, out);
}
